// BDHLanguageModel_48601849921692
// MI455X (gfx1250) — matrix operand order repaired and checked by reading, not compiled, not run on hardware
//
#include <hip/hip_runtime.h>
#include <hip/hip_bf16.h>

#define B_    4
#define T_    512
#define D_    1024
#define V_    32000
#define NSTEP (B_ * T_)            // 2048 sequential steps
#define RBLK  16                   // persistent blocks for recurrence
#define ROWS_PER_BLK (D_ / RBLK)   // 64 rows of W_r per block

typedef __attribute__((ext_vector_type(16))) __bf16 v16bf;
typedef __attribute__((ext_vector_type(8)))  float  v8f;

// ---------------------------------------------------------------------------
// Zero the cross-block barrier counter (workspace is poisoned, not zeroed).
// ---------------------------------------------------------------------------
__global__ void init_ctr_kernel(unsigned int* ctr) { *ctr = 0u; }

// ---------------------------------------------------------------------------
// One-time head_w fp32 -> bf16 conversion (only if workspace is big enough).
// 131 MB read + 65 MB write ~= 8 us at HBM speed; removes 32x-redundant
// per-tile conversion from the GEMM hot loop.
// ---------------------------------------------------------------------------
__global__ __launch_bounds__(256) void convert_w_kernel(
    const float* __restrict__ src, __hip_bfloat16* __restrict__ dst)
{
    const size_t i = ((size_t)blockIdx.x * 256 + threadIdx.x) * 8;  // 8 elems/thread
    float f[8];
    *(float4*)&f[0] = *(const float4*)(src + i);
    *(float4*)&f[4] = *(const float4*)(src + i + 4);
    __hip_bfloat16 b[8];
#pragma unroll
    for (int j = 0; j < 8; ++j) b[j] = __float2bfloat16(f[j]);
    *(uint4*)(dst + i) = *(const uint4*)&b[0];   // 16B store
}

// ---------------------------------------------------------------------------
// Recurrence: v_{s+1} = relu(W_r v_s + E[token_s] + b_r), 2048 sequential steps.
// 16 persistent blocks; each lane holds 64 fp32 of W_r in VGPRs; state is
// broadcast via LDS; blocks sync per step with a release/acquire counter.
// Emits both fp32 states (for the exchange) and bf16 states (GEMM A matrix).
// ---------------------------------------------------------------------------
__global__ __launch_bounds__(1024) void recur_kernel(
    const int*   __restrict__ seq,   // [B, T]
    const float* __restrict__ E,     // [V, D]
    const float* __restrict__ Wr,    // [D, D]
    const float* __restrict__ br,    // [D]
    const float* __restrict__ v0,    // [D]
    float*          __restrict__ vs,    // [NSTEP, D] fp32 states
    __hip_bfloat16* __restrict__ vsbf,  // [NSTEP, D] bf16 states
    unsigned int*   __restrict__ ctr)
{
    __shared__ float v_lds[D_];

    const int tid  = threadIdx.x;
    const int wave = tid >> 5;
    const int lane = tid & 31;
    const int l16  = lane & 15;
    const int h    = lane >> 4;
    const int row  = blockIdx.x * ROWS_PER_BLK + 2 * wave + h;

    // Lane-resident slice of W_r: k = 4*l16 + 64*i + j (bank-conflict-free vs LDS)
    float4 w[16];
    const float* wrow = Wr + (size_t)row * D_;
#pragma unroll
    for (int i = 0; i < 16; ++i)
        w[i] = *(const float4*)(wrow + 4 * l16 + 64 * i);
    const float brv = br[row];

    if (tid < 256)
        *(float4*)&v_lds[tid * 4] = *(const float4*)&v0[tid * 4];
    __syncthreads();

    for (int step = 0; step < NSTEP; ++step) {
        float acc = 0.f;
#pragma unroll
        for (int i = 0; i < 16; ++i) {
            const float4 vv = *(const float4*)&v_lds[4 * l16 + 64 * i];
            acc += w[i].x * vv.x + w[i].y * vv.y + w[i].z * vv.z + w[i].w * vv.w;
        }
        // reduce the 16 lanes of each half-wave (wave32 shuffles)
#pragma unroll
        for (int m = 1; m < 16; m <<= 1)
            acc += __shfl_xor(acc, m, 32);

        if (l16 == 0) {                          // lanes 0 and 16 own rows 2w, 2w+1
            const int t = step >> 2, b = step & 3;   // step = t*B + b
            const int token = seq[b * T_ + t];
            const float x   = E[(size_t)token * D_ + row];
            float val = acc + x + brv;
            val = val > 0.f ? val : 0.f;
            vs[(size_t)step * D_ + row]   = val;
            vsbf[(size_t)step * D_ + row] = __float2bfloat16(val);
        }
        __threadfence();
        __syncthreads();
        if (tid == 0) {
            __hip_atomic_fetch_add(ctr, 1u, __ATOMIC_RELEASE, __HIP_MEMORY_SCOPE_AGENT);
            const unsigned target = (unsigned)(RBLK * (step + 1));
            while (__hip_atomic_load(ctr, __ATOMIC_ACQUIRE, __HIP_MEMORY_SCOPE_AGENT) < target)
                __builtin_amdgcn_s_sleep(2);
        }
        __syncthreads();
        if (tid < 256)
            *(float4*)&v_lds[tid * 4] =
                *(const float4*)&vs[(size_t)step * D_ + tid * 4];
        __syncthreads();
    }
}

// ---------------------------------------------------------------------------
// B-tile loaders (16 contiguous K bf16 per half-wave, same k order as the A tiles).
// ---------------------------------------------------------------------------
__device__ inline v16bf load_b_tile(const __bf16* p) {       // pre-converted path
    union { uint4 u[2]; v16bf v; } c;
    c.u[0] = *(const uint4*)p;
    c.u[1] = *(const uint4*)(p + 8);
    return c.v;
}
__device__ inline v16bf load_b_tile(const float* p) {        // fallback: cvt in-reg
    float f[16];
#pragma unroll
    for (int q = 0; q < 4; ++q)
        *(float4*)&f[q * 4] = *(const float4*)(p + q * 4);
    v16bf v;
#pragma unroll
    for (int j = 0; j < 16; ++j) v[j] = (__bf16)f[j];
    return v;
}

// ---------------------------------------------------------------------------
// Head GEMM: out[b,t,:] = vs[t*B+b,:] @ head_w^T + head_b
// M=2048, N=32000, K=1024. bf16 WMMA, fp32 accumulate.
// 256 threads = 8 waves (2 M x 4 N); wave tile = 32x64 (2x4 WMMA tiles).
// Block tile = 64x256. 8 v_wmma per 12 global_load_b128 per wave per K-step.
// ---------------------------------------------------------------------------
template <typename TB>
__global__ __launch_bounds__(256) void head_gemm_kernel(
    const __bf16* __restrict__ A,    // [NSTEP, D] bf16 states
    const TB*     __restrict__ Wh,   // [V, D] head weight (bf16 or fp32)
    const float*  __restrict__ bh,   // [V]
    float*        __restrict__ out)  // [B, T, V]
{
    const int tid  = threadIdx.x;
    const int wv   = tid >> 5;
    const int lane = tid & 31;
    const int l16  = lane & 15;
    const int h    = lane >> 4;
    const int wm   = wv & 1;          // 2 waves in M
    const int wn   = wv >> 1;         // 4 waves in N
    const int m_base = blockIdx.y * 64  + wm * 32;
    const int n_base = blockIdx.x * 256 + wn * 64;

    v8f acc[2][4] = {};

    for (int kk = 0; kk < D_; kk += 32) {
        // A tiles: 16 contiguous K per half-wave (matches B)
        v16bf a[2];
#pragma unroll
        for (int mi = 0; mi < 2; ++mi) {
            const __bf16* pa = A + (size_t)(m_base + mi * 16 + l16) * D_ + kk;
            union { uint4 u[2]; v16bf v; } ua;
            ua.u[0] = *(const uint4*)(pa + h * 16);
            ua.u[1] = *(const uint4*)(pa + h * 16 + 8);
            a[mi] = ua.v;
        }
        // B tiles: column N = head_w row; 16 contiguous K per half-wave
        v16bf bmat[4];
#pragma unroll
        for (int ni = 0; ni < 4; ++ni) {
            const TB* pb = Wh + (size_t)(n_base + ni * 16 + l16) * D_ + kk + h * 16;
            __builtin_prefetch(pb + 32, 0, 3);   // next K tile; speculative, branch-free
            bmat[ni] = load_b_tile(pb);
        }
#pragma unroll
        for (int mi = 0; mi < 2; ++mi)
#pragma unroll
            for (int ni = 0; ni < 4; ++ni)
                acc[mi][ni] = __builtin_amdgcn_wmma_f32_16x16x32_bf16(
                    false, a[mi], false, bmat[ni],
                    (short)0, acc[mi][ni], false, false);
    }

    // C/D layout: VGPR g -> row M = g + h*8, col N = l16
#pragma unroll
    for (int ni = 0; ni < 4; ++ni) {
        const int col    = n_base + ni * 16 + l16;
        const float bias = bh[col];
#pragma unroll
        for (int mi = 0; mi < 2; ++mi) {
#pragma unroll
            for (int g = 0; g < 8; ++g) {
                const int step = m_base + mi * 16 + g + h * 8;
                const int t = step >> 2, b = step & 3;         // step = t*B + b
                out[((size_t)(b * T_ + t)) * V_ + col] = acc[mi][ni][g] + bias;
            }
        }
    }
}

// ---------------------------------------------------------------------------
extern "C" void kernel_launch(void* const* d_in, const int* in_sizes, int n_in,
                              void* d_out, int out_size, void* d_ws, size_t ws_size,
                              hipStream_t stream) {
    const int*   seq   = (const int*)  d_in[0];
    const float* E     = (const float*)d_in[1];
    const float* Wr    = (const float*)d_in[2];
    const float* br    = (const float*)d_in[3];
    const float* headw = (const float*)d_in[4];
    const float* headb = (const float*)d_in[5];
    const float* v0    = (const float*)d_in[6];
    float* out = (float*)d_out;

    // workspace layout:
    //   [0, 8MB)        vs fp32 states
    //   [8MB, 12MB)     vs bf16 states
    //   [12MB, +256)    barrier counter (+pad)
    //   [12MB+256, ...) head_w bf16 copy (64MB) — only if ws_size allows
    const size_t VS_B   = (size_t)NSTEP * D_ * 4;
    const size_t VSBF_B = (size_t)NSTEP * D_ * 2;
    const size_t CTR_OFF  = VS_B + VSBF_B;
    const size_t WHBF_OFF = CTR_OFF + 256;
    const size_t WHBF_B   = (size_t)V_ * D_ * 2;

    float*          vs   = (float*)d_ws;
    __hip_bfloat16* vsbf = (__hip_bfloat16*)((char*)d_ws + VS_B);
    unsigned int*   ctr  = (unsigned int*)((char*)d_ws + CTR_OFF);
    __hip_bfloat16* whbf = (__hip_bfloat16*)((char*)d_ws + WHBF_OFF);

    const bool preconvert = ws_size >= WHBF_OFF + WHBF_B;

    init_ctr_kernel<<<1, 1, 0, stream>>>(ctr);
    recur_kernel<<<RBLK, 1024, 0, stream>>>(seq, E, Wr, br, v0, vs, vsbf, ctr);

    const dim3 ggrid(V_ / 256, NSTEP / 64);
    if (preconvert) {
        convert_w_kernel<<<((size_t)V_ * D_ / 8 + 255) / 256, 256, 0, stream>>>(headw, whbf);
        head_gemm_kernel<__bf16><<<ggrid, 256, 0, stream>>>(
            (const __bf16*)vsbf, (const __bf16*)whbf, headb, out);
    } else {
        head_gemm_kernel<float><<<ggrid, 256, 0, stream>>>(
            (const __bf16*)vsbf, headw, headb, out);
    }
}
